// MultiHeadAttention_52810917872288
// MI455X (gfx1250) — matrix operand order repaired and checked by reading, not compiled, not run on hardware
//
#include <hip/hip_runtime.h>
#include <hip/hip_bf16.h>

#define DIMSZ   1024
#define NHEADS  16
#define HD      64
#define BATCH   2
#define SEQ     2048
#define ROWS    (BATCH * SEQ)   // 4096
#define QKN     (3 * DIMSZ)     // 3072
#define SPAD    2056            // 2048 + 8 pad, keeps rows 16B-aligned

typedef __bf16 bf16_t;
typedef __attribute__((ext_vector_type(16))) __bf16 v16bf;
typedef __attribute__((ext_vector_type(8)))  float  v8f;
typedef __attribute__((ext_vector_type(4)))  float  f32x4;

union FragBF {
    v16bf v;
    f32x4 q[2];
    unsigned short us[16];
};

__device__ __forceinline__ unsigned short f32_to_bf16(float f) {
    unsigned int u = __float_as_uint(f);
    u += 0x7FFFu + ((u >> 16) & 1u);      // round-to-nearest-even
    return (unsigned short)(u >> 16);
}

__device__ __forceinline__ v8f wmma_bf16(const FragBF& a, const FragBF& b, v8f c) {
    return __builtin_amdgcn_wmma_f32_16x16x32_bf16(
        /*neg_a=*/false, a.v, /*neg_b=*/false, b.v,
        /*c_mod=*/(short)0, c, /*reuse_a=*/false, /*reuse_b=*/false);
}

// A fragment: 16(M) x 32(K), source row-major [rows][ldk] bf16, tile at (row0,k0).
// ISA layout: lane&15 = M row; lanes 0-15 hold K {0..7,16..23}, lanes 16-31 hold K {8..15,24..31}.
__device__ __forceinline__ FragBF load_a_frag(const unsigned short* base, int ldk,
                                              int row0, int k0, int lane) {
    int r = row0 + (lane & 15);
    int half = lane >> 4;
    const unsigned short* p = base + (size_t)r * ldk + k0 + half * 8;
    FragBF f;
    f.q[0] = *(const f32x4*)(p);        // K = k0+half*8 .. +7
    f.q[1] = *(const f32x4*)(p + 16);   // K = k0+16+half*8 .. +7
    return f;
}

// B fragment: 32(K) x 16(N), source stored as N rows x K contiguous (i.e. B^T row-major).
// ISA layout: lane&15 = N col; lanes 0-15 hold K {0..7,16..23}, lanes 16-31 hold K {8..15,24..31}.
__device__ __forceinline__ FragBF load_bt_frag(const unsigned short* base, int ldk,
                                               int n0, int k0, int lane) {
    int n = n0 + (lane & 15);
    int kh = lane >> 4;
    const unsigned short* p = base + (size_t)n * ldk + k0 + kh * 8;
    FragBF f;
    f.q[0] = *(const f32x4*)(p);
    f.q[1] = *(const f32x4*)(p + 16);
    return f;
}

// ---------------- elementwise f32 -> bf16 ----------------
__global__ __launch_bounds__(256)
void cvt_f32_bf16_kernel(const float* __restrict__ in, unsigned short* __restrict__ out, int n) {
    int i = blockIdx.x * 256 + threadIdx.x;
    if (i < n) out[i] = f32_to_bf16(in[i]);
}

// ---------------- V^T builder: vt[bh][d][t] = v[b,t,h*64+d] ----------------
__global__ __launch_bounds__(256)
void build_vt_kernel(const unsigned short* __restrict__ qkv, unsigned short* __restrict__ vt) {
    int idx = blockIdx.x * 256 + threadIdx.x;       // B*H*HD*SEQ = 4194304
    int t  = idx & (SEQ - 1);
    int d  = (idx >> 11) & (HD - 1);
    int bh = idx >> 17;
    int b = bh >> 4, h = bh & 15;
    vt[idx] = qkv[(size_t)(b * SEQ + t) * QKN + 2 * DIMSZ + h * HD + d];
}

// ---------------- gate kernel ----------------
__global__ __launch_bounds__(256)
void gates_kernel(const float* __restrict__ x, const float* __restrict__ gw,
                  const float* __restrict__ gb, const float* __restrict__ gc,
                  float* __restrict__ gates) {
    int idx = blockIdx.x * 256 + threadIdx.x;       // (b*H+h)*SEQ + t, total 65536
    int t = idx & (SEQ - 1);
    int h = (idx >> 11) & 15;
    int b = idx >> 15;
    const float* xi = x + (size_t)(b * SEQ + t) * DIMSZ + h * HD;
    float p[8];
    #pragma unroll
    for (int j = 0; j < 8; ++j) {
        float s = gb[j];
        for (int d = 0; d < HD; ++d) s += xi[d] * gw[j * HD + d];
        p[j] = s;
    }
    float p0 = p[0] + p[1] + p[2] + p[3];
    float p1 = p[4] + p[5] + p[6] + p[7];
    float ga  = 1.0f / (1.0f + __expf(-p0));
    float gbv = 1.0f / (1.0f + __expf(-p1));
    gates[idx] = ga * (gbv * gc[h] - 1.0f) + 2.0f;
}

// ---------------- bf16 WMMA GEMM: C[M,N] = A[M,K] * Bw[N,K]^T + bias ----------------
template <bool OUT_BF16>
__global__ __launch_bounds__(256)
void gemm_bf16_kernel(const unsigned short* __restrict__ A,
                      const unsigned short* __restrict__ Bw,
                      const float* __restrict__ bias,
                      float* __restrict__ outF,
                      unsigned short* __restrict__ outB,
                      int M, int N, int K) {
    int lane = threadIdx.x & 31;
    int w    = threadIdx.x >> 5;
    int wm = w & 3, wn = w >> 2;                  // 4x2 wave grid
    int m0 = blockIdx.y * 128 + wm * 32;          // 2 M-tiles per wave
    int n0 = blockIdx.x * 128 + wn * 64;          // 4 N-tiles per wave

    v8f zero = {0.f, 0.f, 0.f, 0.f, 0.f, 0.f, 0.f, 0.f};
    v8f acc[2][4];
    #pragma unroll
    for (int mi = 0; mi < 2; ++mi)
        #pragma unroll
        for (int nj = 0; nj < 4; ++nj) acc[mi][nj] = zero;

    for (int kc = 0; kc < K; kc += 32) {
        FragBF a0 = load_a_frag(A, K, m0,      kc, lane);
        FragBF a1 = load_a_frag(A, K, m0 + 16, kc, lane);
        FragBF b0 = load_bt_frag(Bw, K, n0,      kc, lane);
        FragBF b1 = load_bt_frag(Bw, K, n0 + 16, kc, lane);
        FragBF b2 = load_bt_frag(Bw, K, n0 + 32, kc, lane);
        FragBF b3 = load_bt_frag(Bw, K, n0 + 48, kc, lane);
        acc[0][0] = wmma_bf16(a0, b0, acc[0][0]);
        acc[0][1] = wmma_bf16(a0, b1, acc[0][1]);
        acc[0][2] = wmma_bf16(a0, b2, acc[0][2]);
        acc[0][3] = wmma_bf16(a0, b3, acc[0][3]);
        acc[1][0] = wmma_bf16(a1, b0, acc[1][0]);
        acc[1][1] = wmma_bf16(a1, b1, acc[1][1]);
        acc[1][2] = wmma_bf16(a1, b2, acc[1][2]);
        acc[1][3] = wmma_bf16(a1, b3, acc[1][3]);
    }

    int half = lane >> 4, nl = lane & 15;
    #pragma unroll
    for (int mi = 0; mi < 2; ++mi)
        #pragma unroll
        for (int nj = 0; nj < 4; ++nj)
            #pragma unroll
            for (int r = 0; r < 8; ++r) {
                int row = m0 + mi * 16 + r + 8 * half;
                int col = n0 + nj * 16 + nl;
                float v = acc[mi][nj][r] + bias[col];
                if (OUT_BF16) outB[(size_t)row * N + col] = f32_to_bf16(v);
                else          outF[(size_t)row * N + col] = v;
            }
}

// ---------------- attention: 32 query rows per block, full score row in 320KB LDS ----------------
__global__ __launch_bounds__(256)
void attn_kernel(const unsigned short* __restrict__ qkv,   // [ROWS][3072] bf16 (q|k|v)
                 const unsigned short* __restrict__ vt,    // [B*H][64][2048] bf16 (V^T)
                 const float* __restrict__ gates,          // [B*H][T]
                 const float* __restrict__ pos_bias,       // [H][T][T]
                 unsigned short* __restrict__ attn_out)    // [ROWS][1024] bf16
{
    extern __shared__ char smem_raw[];
    float* S   = (float*)smem_raw;                           // [32][SPAD]
    float* red = (float*)(smem_raw + (size_t)32 * SPAD * 4); // [32][16]

    int bh = blockIdx.x;
    int b = bh >> 4, h = bh & 15;
    int t0 = blockIdx.y * 32;
    int lane = threadIdx.x & 31;
    int w    = threadIdx.x >> 5;
    int half = lane >> 4, nl = lane & 15;

    // ---- Phase 1: S = Q K^T * scale + gate[t] * pos_bias[h,t,s] ----
    int g  = w & 1;           // row-group (16 rows)
    int m0 = g * 16;
    const unsigned short* qbase = qkv + (size_t)(b * SEQ + t0) * QKN + h * HD;
    FragBF aq0 = load_a_frag(qbase, QKN, m0, 0, lane);
    FragBF aq1 = load_a_frag(qbase, QKN, m0, 32, lane);
    float gt[8];
    #pragma unroll
    for (int r = 0; r < 8; ++r)
        gt[r] = gates[(size_t)bh * SEQ + t0 + m0 + r + 8 * half];

    const unsigned short* kbase = qkv + (size_t)(b * SEQ) * QKN + DIMSZ + h * HD;
    // per-lane base into the bias stream: rows t0+m0+8*half .. step 1 (r), col = s
    const float* pb = pos_bias + (size_t)h * SEQ * SEQ;
    v8f zero = {0.f, 0.f, 0.f, 0.f, 0.f, 0.f, 0.f, 0.f};

    for (int i = 0; i < 32; ++i) {
        int n0 = (w >> 1) * 16 + i * 64;
        FragBF bk0 = load_bt_frag(kbase, QKN, n0, 0, lane);
        FragBF bk1 = load_bt_frag(kbase, QKN, n0, 32, lane);
        v8f acc = zero;
        acc = wmma_bf16(aq0, bk0, acc);
        acc = wmma_bf16(aq1, bk1, acc);
        int s = n0 + nl;

        // Batch the 8 bias loads first so they clause under a single wait
        // (this is the 512MB HBM stream that sets the roofline).
        const float* pbl = pb + (size_t)(t0 + m0 + 8 * half) * SEQ + s;
        float bias_v[8];
        #pragma unroll
        for (int r = 0; r < 8; ++r) bias_v[r] = pbl[(size_t)r * SEQ];

        #pragma unroll
        for (int r = 0; r < 8; ++r) {
            int trow = m0 + r + 8 * half;                  // 0..31 within block
            float v = acc[r] * 0.125f + gt[r] * bias_v[r];
            S[(size_t)trow * SPAD + s] = v;
        }
    }
    __syncthreads();

    // ---- Phase 2: row softmax (store un-normalized exp + 1/rowsum) ----
    {
        int r   = threadIdx.x >> 3;    // 0..31
        int seg = threadIdx.x & 7;     // 0..7, 256 cols each
        float* Sr = S + (size_t)r * SPAD + seg * 256;
        float lmax = -3.0e38f;
        for (int c = 0; c < 256; ++c) lmax = fmaxf(lmax, Sr[c]);
        red[r * 16 + seg] = lmax;
        __syncthreads();
        float rmax = red[r * 16 + 0];
        #pragma unroll
        for (int j = 1; j < 8; ++j) rmax = fmaxf(rmax, red[r * 16 + j]);
        __syncthreads();
        float lsum = 0.f;
        for (int c = 0; c < 256; ++c) {
            float e = __expf(Sr[c] - rmax);
            Sr[c] = e;
            lsum += e;
        }
        red[r * 16 + seg] = lsum;
        __syncthreads();
        if (seg == 0) {
            float rs = 0.f;
            #pragma unroll
            for (int j = 0; j < 8; ++j) rs += red[r * 16 + j];
            red[r * 16 + 8] = 1.0f / rs;
        }
        __syncthreads();
    }

    // ---- Phase 3: O = P V (normalize in epilogue) ----
    int m02 = (w & 1) * 16;
    int d0  = (w >> 1) * 16;
    const unsigned short* vtb = vt + (size_t)bh * HD * SEQ;  // rows = d, ld = SEQ
    v8f acc = zero;
    for (int kc = 0; kc < SEQ; kc += 32) {
        // A fragment straight out of LDS as 4x ds_load_b128, converted to bf16.
        const float* sp = S + (size_t)(m02 + nl) * SPAD + kc + half * 8;
        f32x4 x0 = *(const f32x4*)(sp);
        f32x4 x1 = *(const f32x4*)(sp + 4);
        f32x4 x2 = *(const f32x4*)(sp + 16);
        f32x4 x3 = *(const f32x4*)(sp + 20);
        FragBF ap;
        #pragma unroll
        for (int e = 0; e < 4; ++e) {
            ap.us[e]      = f32_to_bf16(x0[e]);
            ap.us[4 + e]  = f32_to_bf16(x1[e]);
            ap.us[8 + e]  = f32_to_bf16(x2[e]);
            ap.us[12 + e] = f32_to_bf16(x3[e]);
        }
        FragBF bv = load_bt_frag(vtb, SEQ, d0, kc, lane);
        acc = wmma_bf16(ap, bv, acc);
    }
    #pragma unroll
    for (int r = 0; r < 8; ++r) {
        int mrow = m02 + r + 8 * half;
        float inv = red[mrow * 16 + 8];
        int t = t0 + mrow;
        int d = d0 + nl;
        attn_out[(size_t)(b * SEQ + t) * DIMSZ + h * HD + d] = f32_to_bf16(acc[r] * inv);
    }
}

extern "C" void kernel_launch(void* const* d_in, const int* in_sizes, int n_in,
                              void* d_out, int out_size, void* d_ws, size_t ws_size,
                              hipStream_t stream) {
    const float* input    = (const float*)d_in[0];
    const float* pos_bias = (const float*)d_in[1];
    const float* qkv_w    = (const float*)d_in[2];
    const float* qkv_b    = (const float*)d_in[3];
    const float* out_w    = (const float*)d_in[4];
    const float* out_b    = (const float*)d_in[5];
    const float* gru_w    = (const float*)d_in[6];
    const float* gru_b    = (const float*)d_in[7];
    const float* gru_c    = (const float*)d_in[8];
    float* out = (float*)d_out;

    char* ws = (char*)d_ws;
    size_t off = 0;
    auto alloc = [&](size_t bytes) {
        char* p = ws + off;
        off += (bytes + 255) & ~(size_t)255;
        return p;
    };
    unsigned short* xb    = (unsigned short*)alloc((size_t)ROWS * DIMSZ * 2);   // input bf16
    unsigned short* wqkvb = (unsigned short*)alloc((size_t)QKN * DIMSZ * 2);    // qkv_w bf16
    unsigned short* owb   = (unsigned short*)alloc((size_t)DIMSZ * DIMSZ * 2);  // out_w bf16
    unsigned short* qkvb  = (unsigned short*)alloc((size_t)ROWS * QKN * 2);     // qkv bf16
    unsigned short* vtb   = (unsigned short*)alloc((size_t)BATCH * NHEADS * HD * SEQ * 2);
    float*          gts   = (float*)alloc((size_t)BATCH * NHEADS * SEQ * 4);
    unsigned short* aob   = (unsigned short*)alloc((size_t)ROWS * DIMSZ * 2);   // attn out bf16

    // 1) conversions
    cvt_f32_bf16_kernel<<<(ROWS * DIMSZ) / 256, 256, 0, stream>>>(input, xb, ROWS * DIMSZ);
    cvt_f32_bf16_kernel<<<(QKN * DIMSZ) / 256, 256, 0, stream>>>(qkv_w, wqkvb, QKN * DIMSZ);
    cvt_f32_bf16_kernel<<<(DIMSZ * DIMSZ) / 256, 256, 0, stream>>>(out_w, owb, DIMSZ * DIMSZ);

    // 2) gates
    gates_kernel<<<(BATCH * NHEADS * SEQ) / 256, 256, 0, stream>>>(input, gru_w, gru_b, gru_c, gts);

    // 3) QKV projection (bf16 out, bias fused)
    gemm_bf16_kernel<true><<<dim3(QKN / 128, ROWS / 128), 256, 0, stream>>>(
        xb, wqkvb, qkv_b, nullptr, qkvb, ROWS, QKN, DIMSZ);

    // 4) V^T layout for the PV GEMM
    build_vt_kernel<<<(BATCH * NHEADS * HD * SEQ) / 256, 256, 0, stream>>>(qkvb, vtb);

    // 5) attention (32 query rows per block; ~259KB dynamic LDS of the 320KB WGP pool)
    size_t smem = (size_t)32 * SPAD * 4 + (size_t)32 * 16 * 4;
    attn_kernel<<<dim3(BATCH * NHEADS, SEQ / 32), 256, smem, stream>>>(
        qkvb, vtb, gts, pos_bias, aob);

    // 6) output projection (f32 out, bias fused)
    gemm_bf16_kernel<false><<<dim3(DIMSZ / 128, ROWS / 128), 256, 0, stream>>>(
        aob, owb, out_b, out, nullptr, ROWS, DIMSZ, DIMSZ);
}
